// Model_30477087933020
// MI455X (gfx1250) — matrix operand order repaired and checked by reading, not compiled, not run on hardware
//
#include <hip/hip_runtime.h>
#include <hip/hip_bf16.h>

#define DIM 64
#define LSTR 72   // padded row stride (ushorts) for transposed W in LDS:
                  // 144B/row -> fragment bases stay 16B-aligned, banks spread

typedef __attribute__((ext_vector_type(16))) __bf16         v16bf;
typedef __attribute__((ext_vector_type(16))) unsigned short v16u;
typedef __attribute__((ext_vector_type(8)))  float          v8f;

// Round-to-nearest-even f32 -> bf16 bits, plus residual ("bf16x3" split):
// f = hi + lo; hi*hi + hi*lo + lo*hi recovers ~fp32 accuracy.
__device__ inline void bf_split_bits(float f, unsigned short& hi, unsigned short& lo) {
    unsigned u = __float_as_uint(f);
    unsigned r = u + 0x7FFFu + ((u >> 16) & 1u);
    hi = (unsigned short)(r >> 16);
    float fh = __uint_as_float(((unsigned)hi) << 16);
    float res = f - fh;                       // exact in f32
    unsigned u2 = __float_as_uint(res);
    unsigned r2 = u2 + 0x7FFFu + ((u2 >> 16) & 1u);
    lo = (unsigned short)(r2 >> 16);
}

// ---------------- degree / normalization ----------------

__global__ void gcn_deg_init(float* __restrict__ deg, int N) {
    int i = blockIdx.x * blockDim.x + threadIdx.x;
    if (i < N) deg[i] = 1.0f;                 // self-loop contributes 1
}

__global__ void gcn_deg_count(const long long* __restrict__ ei,
                              float* __restrict__ deg, int E) {
    int e = blockIdx.x * blockDim.x + threadIdx.x;
    if (e < E) {
        int dst = (int)ei[(long long)E + e];  // row 1 of edge_index
        unsafeAtomicAdd(&deg[dst], 1.0f);     // native global_atomic_add_f32
    }
}

__global__ void gcn_deg_rsqrt(float* __restrict__ deg, int N) {
    int i = blockIdx.x * blockDim.x + threadIdx.x;
    if (i < N) deg[i] = rsqrtf(deg[i]);       // deg >= 1 always (self-loops)
}

// ---------------- GEMM: T[N x 64] = H[N x 64] @ W[64 x 64] ----------------
// 256 threads = 8 waves; each wave owns a 16-row tile (128 rows / block).
// W staged in LDS transposed (n-major) so each B fragment is one contiguous
// 32B vector read per lane. bf16x3 split: 24 v_wmma per wave.

__global__ __launch_bounds__(256) void gcn_gemm(const float* __restrict__ H,
                                                const float* __restrict__ W,
                                                float* __restrict__ T, int N) {
    __shared__ __align__(16) unsigned short Wth[DIM * LSTR]; // bf16 hi, [n][k]
    __shared__ __align__(16) unsigned short Wtl[DIM * LSTR]; // bf16 lo, [n][k]

    const int tid = threadIdx.x;
    for (int i = tid; i < DIM * DIM; i += 256) {
        const int k = i >> 6, n = i & 63;     // W is row-major [k][n]
        unsigned short hi, lo; bf_split_bits(W[i], hi, lo);
        Wth[n * LSTR + k] = hi;               // transpose at store time
        Wtl[n * LSTR + k] = lo;
    }
    __syncthreads();

    const int wave = tid >> 5;
    const int lane = tid & 31;
    const int grp  = lane >> 4;               // 0: lanes 0-15, 1: lanes 16-31
    const int l16  = lane & 15;
    const int row0 = blockIdx.x * 128 + wave * 16;
    const int m    = row0 + l16;              // A-matrix row this lane feeds

    v8f acc[4] = {};                          // four 16x16 output col-tiles

    #pragma unroll
    for (int kb = 0; kb < DIM; kb += 32) {
        // A fragment (16x32 bf16): lane holds K = kb+grp*16+0..15
        v16u auh, aul;
        #pragma unroll
        for (int half = 0; half < 2; ++half) {
            const int kbase = kb + grp * 16 + half * 8;   // 8 contiguous floats
            float4 q0 = make_float4(0.f, 0.f, 0.f, 0.f);
            float4 q1 = make_float4(0.f, 0.f, 0.f, 0.f);
            if (m < N) {
                q0 = *(const float4*)&H[m * DIM + kbase];
                q1 = *(const float4*)&H[m * DIM + kbase + 4];
            }
            const float f[8] = {q0.x, q0.y, q0.z, q0.w, q1.x, q1.y, q1.z, q1.w};
            #pragma unroll
            for (int t = 0; t < 8; ++t) {
                unsigned short hi, lo; bf_split_bits(f[t], hi, lo);
                auh[half * 8 + t] = hi;
                aul[half * 8 + t] = lo;
            }
        }
        const v16bf ahi = __builtin_bit_cast(v16bf, auh);
        const v16bf alo = __builtin_bit_cast(v16bf, aul);

        #pragma unroll
        for (int ct = 0; ct < 4; ++ct) {
            const int n   = ct * 16 + l16;    // B col this lane feeds
            const int off = n * LSTR + kb + grp * 16;     // 16B-aligned
            const v16bf bhi = __builtin_bit_cast(v16bf, *(const v16u*)&Wth[off]);
            const v16bf blo = __builtin_bit_cast(v16bf, *(const v16u*)&Wtl[off]);
            acc[ct] = __builtin_amdgcn_wmma_f32_16x16x32_bf16(
                false, ahi, false, bhi, (short)0, acc[ct], false, false);
            acc[ct] = __builtin_amdgcn_wmma_f32_16x16x32_bf16(
                false, ahi, false, blo, (short)0, acc[ct], false, false);
            acc[ct] = __builtin_amdgcn_wmma_f32_16x16x32_bf16(
                false, alo, false, bhi, (short)0, acc[ct], false, false);
        }
    }

    // C/D layout: VGPR v -> row v (lanes 0-15) or v+8 (lanes 16-31); col = lane&15
    #pragma unroll
    for (int ct = 0; ct < 4; ++ct) {
        #pragma unroll
        for (int v = 0; v < 8; ++v) {
            const int r = row0 + grp * 8 + v;
            if (r < N) T[r * DIM + ct * 16 + l16] = acc[ct][v];
        }
    }
}

// ---------------- aggregation ----------------

// A[i][:] = dis[i]^2 * T[i][:]   (self-loop term; also zero-inits A)
__global__ void gcn_agg_init(const float* __restrict__ T,
                             const float* __restrict__ dis,
                             float* __restrict__ A, int N) {
    int i = blockIdx.x * blockDim.x + threadIdx.x;
    if (i < N * DIM) {
        float s = dis[i >> 6];
        A[i] = s * s * T[i];
    }
}

// 64 consecutive threads per edge -> coalesced gather + coalesced atomics
__global__ void gcn_scatter(const long long* __restrict__ ei,
                            const float* __restrict__ T,
                            const float* __restrict__ dis,
                            float* __restrict__ A, int E) {
    long long gid = (long long)blockIdx.x * blockDim.x + threadIdx.x;
    int e = (int)(gid >> 6);
    int d = (int)(gid & 63);
    if (e < E) {
        int s = (int)ei[e];
        int t = (int)ei[(long long)E + e];
        float nrm = dis[s] * dis[t];
        unsafeAtomicAdd(&A[t * DIM + d], nrm * T[s * DIM + d]);
    }
}

__global__ void gcn_bias_act(float* __restrict__ A, const float* __restrict__ b,
                             int N, int do_relu) {
    int i = blockIdx.x * blockDim.x + threadIdx.x;
    if (i < N * DIM) {
        float v = A[i] + b[i & 63];
        A[i] = do_relu ? fmaxf(v, 0.0f) : v;
    }
}

// ---------------- driver ----------------

extern "C" void kernel_launch(void* const* d_in, const int* in_sizes, int n_in,
                              void* d_out, int out_size, void* d_ws, size_t ws_size,
                              hipStream_t stream) {
    const float*     x  = (const float*)d_in[0];
    const long long* ei = (const long long*)d_in[1];   // int64 edge_index [2,E]
    const float*     W1 = (const float*)d_in[2];
    const float*     b1 = (const float*)d_in[3];
    const float*     W2 = (const float*)d_in[4];
    const float*     b2 = (const float*)d_in[5];
    const float*     W3 = (const float*)d_in[6];
    const float*     b3 = (const float*)d_in[7];
    float* out = (float*)d_out;

    const int N = in_sizes[0] / DIM;     // 100000
    const int E = in_sizes[1] / 2;       // 1600000

    // workspace: dis[N] | T[N*64] | Hb[N*64]   (~52 MB)
    float* dis = (float*)d_ws;
    float* T   = dis + (((size_t)N + 255) & ~(size_t)255);
    float* Hb  = T + (size_t)N * DIM;

    const int nb  = (N + 255) / 256;
    const int ebk = (E + 255) / 256;
    const int gbk = (N + 127) / 128;                       // GEMM blocks
    const int fbk = (int)(((long long)N * DIM + 255) / 256);
    const int sbk = (int)(((long long)E * DIM + 255) / 256);

    // degrees -> dis = rsqrt(deg)
    gcn_deg_init <<<nb,  256, 0, stream>>>(dis, N);
    gcn_deg_count<<<ebk, 256, 0, stream>>>(ei, dis, E);
    gcn_deg_rsqrt<<<nb,  256, 0, stream>>>(dis, N);

    // layer 1: d_out = relu(agg(x@W1) + b1)
    gcn_gemm     <<<gbk, 256, 0, stream>>>(x, W1, T, N);
    gcn_agg_init <<<fbk, 256, 0, stream>>>(T, dis, out, N);
    gcn_scatter  <<<sbk, 256, 0, stream>>>(ei, T, dis, out, E);
    gcn_bias_act <<<fbk, 256, 0, stream>>>(out, b1, N, 1);

    // layer 2: Hb = relu(agg(d_out@W2) + b2)
    gcn_gemm     <<<gbk, 256, 0, stream>>>(out, W2, T, N);
    gcn_agg_init <<<fbk, 256, 0, stream>>>(T, dis, Hb, N);
    gcn_scatter  <<<sbk, 256, 0, stream>>>(ei, T, dis, Hb, E);
    gcn_bias_act <<<fbk, 256, 0, stream>>>(Hb, b2, N, 1);

    // layer 3: d_out = agg(Hb@W3) + b3
    gcn_gemm     <<<gbk, 256, 0, stream>>>(Hb, W3, T, N);
    gcn_agg_init <<<fbk, 256, 0, stream>>>(T, dis, out, N);
    gcn_scatter  <<<sbk, 256, 0, stream>>>(ei, T, dis, out, E);
    gcn_bias_act <<<fbk, 256, 0, stream>>>(out, b3, N, 0);
}
